// Transformer_learned_trig_31550829756766
// MI455X (gfx1250) — matrix operand order repaired and checked by reading, not compiled, not run on hardware
//
#include <hip/hip_runtime.h>
#include <math.h>

// ---------------------------------------------------------------------------
// Performer linear-attention forward for MI455X (gfx1250, wave32).
// All large GEMMs use v_wmma_f32_16x16x32_bf16 (bf16 operands, f32 accum).
// Problem is HBM-bandwidth dominated (~170MB unique traffic vs 30 GFLOP),
// so intermediates are bf16 and stored in WMMA-operand-friendly layouts.
// ---------------------------------------------------------------------------

typedef __bf16 v16bf __attribute__((ext_vector_type(16)));
typedef __bf16 v8bf  __attribute__((ext_vector_type(8)));
typedef float  v8f   __attribute__((ext_vector_type(8)));

#define NB   8
#define NL   2048
#define DLEN 512
#define DM   256
#define MF   256
#define NC   64
#define ROWS (NB * NL)          // 16384
#define TWO_PI_F 6.28318530717958647692f

// ---- workspace layout (bytes) ---------------------------------------------
#define OFF_XBF   ((size_t)0)                                   // 16384*512*2
#define OFF_WQB   (OFF_XBF  + (size_t)ROWS * DLEN * 2)
#define OFF_WKB   (OFF_WQB  + (size_t)DM * DLEN * 2)
#define OFF_WVB   (OFF_WKB  + (size_t)DM * DLEN * 2)
#define OFF_WAB   (OFF_WVB  + (size_t)DM * DLEN * 2)
#define OFF_RFB   (OFF_WAB  + (size_t)DM * DLEN * 2)
#define OFF_QB    (OFF_RFB  + (size_t)MF * DM * 2)
#define OFF_KB    (OFF_QB   + (size_t)ROWS * DM * 2)
#define OFF_VT    (OFF_KB   + (size_t)ROWS * DM * 2)            // [b][d][l]
#define OFF_ADD   (OFF_VT   + (size_t)ROWS * DM * 2)            // f32
#define OFF_HQ    (OFF_ADD  + (size_t)ROWS * DM * 4)
#define OFF_HK    (OFF_HQ   + (size_t)ROWS * 4)
#define OFF_QP    (OFF_HK   + (size_t)ROWS * 4)                 // [row][2m]
#define OFF_KPT   (OFF_QP   + (size_t)ROWS * 2 * MF * 2)        // [b][2m][l]
#define OFF_KSUM  (OFF_KPT  + (size_t)ROWS * 2 * MF * 2)        // f32 [b][2m]
#define OFF_DEN   (OFF_KSUM + (size_t)NB * 2 * MF * 4)          // f32 [row]
#define OFF_KVT   (OFF_DEN  + (size_t)ROWS * 4)                 // [b][d][2m]
#define OFF_OUT1  (OFF_KVT  + (size_t)NB * DM * 2 * MF * 2)     // f32
#define OFF_PART  (OFF_OUT1 + (size_t)ROWS * DM * 4)            // f32 [8][64][8]

// ---- helpers ---------------------------------------------------------------
__device__ __forceinline__ unsigned short f2bf(float f) {
  unsigned int u = __float_as_uint(f);
  u += 0x7FFFu + ((u >> 16) & 1u);          // round-to-nearest-even
  return (unsigned short)(u >> 16);
}
__device__ __forceinline__ float bf2f(unsigned short h) {
  return __uint_as_float(((unsigned int)h) << 16);
}

// A operand: 16x32 bf16 tile. Lane layout: lane g=lane>>4,
// m=lane&15; elems 0..15 -> K = k0+16g .. k0+16g+15 contiguous.
__device__ __forceinline__ v16bf load_a(const unsigned short* rowptr, int k0, int g) {
  return *(const v16bf*)(rowptr + k0 + 16 * g);
}
// B operand: 32x16 bf16 tile, source stores B^T rows (N-major, K contiguous).
// Lane n=lane&15 holds K = k0+16g .. k0+16g+15 contiguous.
__device__ __forceinline__ v16bf load_b(const unsigned short* colptr, int k0, int g) {
  return *(const v16bf*)(colptr + k0 + 16 * g);
}
__device__ __forceinline__ v8f wmma_bf16(v16bf a, v16bf b, v8f c) {
  return __builtin_amdgcn_wmma_f32_16x16x32_bf16(false, a, false, b,
                                                 (short)0, c, false, false);
}
// Pack 8 f32 accumulator rows into 8 contiguous bf16 (16B store).
__device__ __forceinline__ void store_bf8(unsigned short* p, const float v[8]) {
  uint4 pk;
  pk.x = (unsigned int)f2bf(v[0]) | ((unsigned int)f2bf(v[1]) << 16);
  pk.y = (unsigned int)f2bf(v[2]) | ((unsigned int)f2bf(v[3]) << 16);
  pk.z = (unsigned int)f2bf(v[4]) | ((unsigned int)f2bf(v[5]) << 16);
  pk.w = (unsigned int)f2bf(v[6]) | ((unsigned int)f2bf(v[7]) << 16);
  *(uint4*)p = pk;
}

// ---- kernels ---------------------------------------------------------------
__global__ void k_f32_to_bf16(const float* __restrict__ src,
                              unsigned short* __restrict__ dst, int n) {
  for (int i = blockIdx.x * blockDim.x + threadIdx.x; i < n;
       i += gridDim.x * blockDim.x)
    dst[i] = f2bf(src[i]);
}

// Fused Q/K/V/W_add projection: X[16384,512] x W^T -> [16384,256] each.
// grid (16 col tiles, 1024 row tiles), block = one wave32.
__global__ void k_qkv(const unsigned short* __restrict__ X,
                      const unsigned short* __restrict__ Wqb,
                      const unsigned short* __restrict__ Wkb,
                      const unsigned short* __restrict__ Wvb,
                      const unsigned short* __restrict__ Wab,
                      unsigned short* __restrict__ Qb,
                      unsigned short* __restrict__ Kb,
                      unsigned short* __restrict__ Vt,
                      float* __restrict__ Add) {
  const int lane = threadIdx.x & 31;
  const int g = lane >> 4, nm = lane & 15;
  const int col0 = blockIdx.x * 16;
  const int row0 = blockIdx.y * 16;
  const unsigned short* xrow = X + (size_t)(row0 + nm) * DLEN;
  const unsigned short* wq = Wqb + (size_t)(col0 + nm) * DLEN;
  const unsigned short* wk = Wkb + (size_t)(col0 + nm) * DLEN;
  const unsigned short* wv = Wvb + (size_t)(col0 + nm) * DLEN;
  const unsigned short* wa = Wab + (size_t)(col0 + nm) * DLEN;
  v8f aq = {0,0,0,0,0,0,0,0}, ak = aq, av = aq, aa = aq;
  for (int k0 = 0; k0 < DLEN; k0 += 32) {
    v16bf a = load_a(xrow, k0, g);
    aq = wmma_bf16(a, load_b(wq, k0, g), aq);
    ak = wmma_bf16(a, load_b(wk, k0, g), ak);
    av = wmma_bf16(a, load_b(wv, k0, g), av);
    aa = wmma_bf16(a, load_b(wa, k0, g), aa);
  }
  const int col = col0 + nm;
  float vv[8];
#pragma unroll
  for (int r = 0; r < 8; ++r) {
    const int row = row0 + r + 8 * g;
    Qb[(size_t)row * DM + col] = f2bf(aq[r]);
    Kb[(size_t)row * DM + col] = f2bf(ak[r]);
    Add[(size_t)row * DM + col] = aa[r];
    vv[r] = av[r];
  }
  const int b = row0 >> 11, lloc = (row0 & (NL - 1)) + 8 * g;
  store_bf8(Vt + ((size_t)b * DM + col) * NL + lloc, vv);   // V^T [b][d][l]
}

// Row norms -> h = exp(0.5*||row||^2)/sqrt(M). grid (2048, 2) block 256.
__global__ void k_norm(const unsigned short* __restrict__ Qb,
                       const unsigned short* __restrict__ Kb,
                       float* __restrict__ Hq, float* __restrict__ Hk) {
  const unsigned short* src = blockIdx.y ? Kb : Qb;
  float* dst = blockIdx.y ? Hk : Hq;
  const int warp = threadIdx.x >> 5, lane = threadIdx.x & 31;
  const int row = blockIdx.x * 8 + warp;
  const unsigned short* p = src + (size_t)row * DM;
  float s = 0.f;
#pragma unroll
  for (int j = 0; j < 8; ++j) { float v = bf2f(p[lane + 32 * j]); s += v * v; }
#pragma unroll
  for (int off = 16; off > 0; off >>= 1) s += __shfl_down(s, off, 32);
  if (lane == 0) dst[row] = expf(0.5f * s) * 0.0625f;   // 1/sqrt(256)
}

// phi: proj = (Q|K) @ RF^T via WMMA, then h*{sin,cos}(2*pi*proj).
// q' row-major [row][2m]; k' transposed [b][2m][l]. grid (16,1024,2) blk 32.
__global__ void k_phi(const unsigned short* __restrict__ Qb,
                      const unsigned short* __restrict__ Kb,
                      const unsigned short* __restrict__ RF,
                      const float* __restrict__ Hq, const float* __restrict__ Hk,
                      unsigned short* __restrict__ qp,
                      unsigned short* __restrict__ kpT) {
  const int lane = threadIdx.x & 31;
  const int g = lane >> 4, nm = lane & 15;
  const int col0 = blockIdx.x * 16;   // random-feature index m
  const int row0 = blockIdx.y * 16;   // token row
  const int isK = blockIdx.z;
  const unsigned short* src = isK ? Kb : Qb;
  const float* H = isK ? Hk : Hq;
  const unsigned short* arow = src + (size_t)(row0 + nm) * DM;
  const unsigned short* brow = RF + (size_t)(col0 + nm) * DM;
  v8f acc = {0,0,0,0,0,0,0,0};
  for (int k0 = 0; k0 < DM; k0 += 32)
    acc = wmma_bf16(load_a(arow, k0, g), load_b(brow, k0, g), acc);
  const int col = col0 + nm;
  float sv[8], cv[8];
#pragma unroll
  for (int r = 0; r < 8; ++r) {
    const int row = row0 + r + 8 * g;
    const float ang = TWO_PI_F * acc[r];
    const float h = H[row];
    sv[r] = h * sinf(ang);
    cv[r] = h * cosf(ang);
  }
  if (!isK) {
#pragma unroll
    for (int r = 0; r < 8; ++r) {
      const int row = row0 + r + 8 * g;
      qp[(size_t)row * (2 * MF) + col]      = f2bf(sv[r]);
      qp[(size_t)row * (2 * MF) + MF + col] = f2bf(cv[r]);
    }
  } else {
    const int b = row0 >> 11, lloc = (row0 & (NL - 1)) + 8 * g;
    store_bf8(kpT + ((size_t)b * 2 * MF + col) * NL + lloc, sv);
    store_bf8(kpT + ((size_t)b * 2 * MF + MF + col) * NL + lloc, cv);
  }
}

// k_sum[b][m] = sum_l k'[b][l][m] from kpT rows. grid 4096, block 128.
__global__ void k_ksum(const unsigned short* __restrict__ kpT,
                       float* __restrict__ ksum) {
  __shared__ float red[128];
  const int bm = blockIdx.x;                       // b*512 + m
  const unsigned short* p = kpT + (size_t)bm * NL;
  float s = 0.f;
  for (int i = threadIdx.x; i < NL; i += 128) s += bf2f(p[i]);
  red[threadIdx.x] = s;
  __syncthreads();
  for (int off = 64; off > 0; off >>= 1) {
    if (threadIdx.x < off) red[threadIdx.x] += red[threadIdx.x + off];
    __syncthreads();
  }
  if (threadIdx.x == 0) ksum[bm] = red[0];
}

// kv[b][m][d] = sum_l k'[l][m] v[l][d]; both operands contiguous (kpT, vT),
// output stored transposed kvT [b][d][2m]. grid (16 d, 32 m, 8 b) block 32.
__global__ void k_kv(const unsigned short* __restrict__ kpT,
                     const unsigned short* __restrict__ vT,
                     unsigned short* __restrict__ kvT) {
  const int lane = threadIdx.x & 31;
  const int g = lane >> 4, nm = lane & 15;
  const int b = blockIdx.z;
  const int m0 = blockIdx.y * 16, d0 = blockIdx.x * 16;
  const unsigned short* arow = kpT + ((size_t)b * 2 * MF + m0 + nm) * NL;
  const unsigned short* brow = vT + ((size_t)b * DM + d0 + nm) * NL;
  v8f acc = {0,0,0,0,0,0,0,0};
  for (int k0 = 0; k0 < NL; k0 += 32)
    acc = wmma_bf16(load_a(arow, k0, g), load_b(brow, k0, g), acc);
  float vv[8];
#pragma unroll
  for (int r = 0; r < 8; ++r) vv[r] = acc[r];
  // lane holds column d = d0+nm, rows m = m0 + r + 8g -> contiguous in kvT
  store_bf8(kvT + ((size_t)b * DM + d0 + nm) * (2 * MF) + m0 + 8 * g, vv);
}

// denom[row] = q'[row] . ksum[b]. grid 2048, block 256 (warp per row).
__global__ void k_denom(const unsigned short* __restrict__ qp,
                        const float* __restrict__ ksum,
                        float* __restrict__ denom) {
  const int warp = threadIdx.x >> 5, lane = threadIdx.x & 31;
  const int row = blockIdx.x * 8 + warp;
  const int b = row >> 11;
  const unsigned short* q = qp + (size_t)row * (2 * MF);
  const float* ks = ksum + (size_t)b * 2 * MF;
  float s = 0.f;
  for (int m = lane; m < 2 * MF; m += 32) s += bf2f(q[m]) * ks[m];
#pragma unroll
  for (int off = 16; off > 0; off >>= 1) s += __shfl_down(s, off, 32);
  if (lane == 0) denom[row] = s;
}

// out1 = (q' @ kv)/denom + Add + b_add. grid (16,1024) block 32.
__global__ void k_out1(const unsigned short* __restrict__ qp,
                       const unsigned short* __restrict__ kvT,
                       const float* __restrict__ denom,
                       const float* __restrict__ Add,
                       const float* __restrict__ b_add,
                       float* __restrict__ out1) {
  const int lane = threadIdx.x & 31;
  const int g = lane >> 4, nm = lane & 15;
  const int d0 = blockIdx.x * 16;
  const int row0 = blockIdx.y * 16;
  const int b = row0 >> 11;
  const unsigned short* arow = qp + (size_t)(row0 + nm) * (2 * MF);
  const unsigned short* brow = kvT + ((size_t)b * DM + d0 + nm) * (2 * MF);
  v8f acc = {0,0,0,0,0,0,0,0};
  for (int k0 = 0; k0 < 2 * MF; k0 += 32)
    acc = wmma_bf16(load_a(arow, k0, g), load_b(brow, k0, g), acc);
  const int col = d0 + nm;
  const float bb = b_add[col];
#pragma unroll
  for (int r = 0; r < 8; ++r) {
    const int row = row0 + r + 8 * g;
    out1[(size_t)row * DM + col] =
        acc[r] / denom[row] + Add[(size_t)row * DM + col] + bb;
  }
}

// Split-K partial for final layer: W_final read exactly once.
// grid (8 k-chunks, 64 classes), block 256. K chunk = 65536.
__global__ void k_final_partial(const float* __restrict__ out1,
                                const float* __restrict__ Wf,
                                float* __restrict__ partial) {
  __shared__ float red[256];
  const int kc = blockIdx.x, c = blockIdx.y;
  const size_t base = (size_t)kc * 65536;
  const float* w = Wf + (size_t)c * (ROWS / NB * DM) + base;   // 524288 stride
  float acc[NB] = {0,0,0,0,0,0,0,0};
  for (int i = threadIdx.x; i < 65536; i += 256) {
    const float wv = w[i];
#pragma unroll
    for (int b = 0; b < NB; ++b)
      acc[b] += wv * out1[(size_t)b * 524288 + base + i];
  }
#pragma unroll 1
  for (int b = 0; b < NB; ++b) {
    red[threadIdx.x] = acc[b];
    __syncthreads();
    for (int off = 128; off > 0; off >>= 1) {
      if (threadIdx.x < off) red[threadIdx.x] += red[threadIdx.x + off];
      __syncthreads();
    }
    if (threadIdx.x == 0) partial[((size_t)kc * NC + c) * NB + b] = red[0];
    __syncthreads();
  }
}

// Sum 8 partials per (b,c), add b_final. 1 block, 512 threads.
__global__ void k_final_reduce(const float* __restrict__ partial,
                               const float* __restrict__ b_final,
                               float* __restrict__ out) {
  const int t = threadIdx.x;          // t = b*64 + c
  const int b = t >> 6, c = t & 63;
  float s = b_final[c];
#pragma unroll
  for (int kc = 0; kc < 8; ++kc) s += partial[((size_t)kc * NC + c) * NB + b];
  out[t] = s;
}

// ---------------------------------------------------------------------------
extern "C" void kernel_launch(void* const* d_in, const int* in_sizes, int n_in,
                              void* d_out, int out_size, void* d_ws, size_t ws_size,
                              hipStream_t stream) {
  (void)in_sizes; (void)n_in; (void)out_size; (void)ws_size;
  const float* x     = (const float*)d_in[0];
  const float* Wq    = (const float*)d_in[7];
  const float* Wk    = (const float*)d_in[8];
  const float* Wv    = (const float*)d_in[9];
  const float* Wadd  = (const float*)d_in[10];
  const float* b_add = (const float*)d_in[11];
  const float* Wfin  = (const float*)d_in[12];
  const float* b_fin = (const float*)d_in[13];
  const float* rf    = (const float*)d_in[14];

  char* ws = (char*)d_ws;
  unsigned short* XBF  = (unsigned short*)(ws + OFF_XBF);
  unsigned short* WQB  = (unsigned short*)(ws + OFF_WQB);
  unsigned short* WKB  = (unsigned short*)(ws + OFF_WKB);
  unsigned short* WVB  = (unsigned short*)(ws + OFF_WVB);
  unsigned short* WAB  = (unsigned short*)(ws + OFF_WAB);
  unsigned short* RFB  = (unsigned short*)(ws + OFF_RFB);
  unsigned short* QB   = (unsigned short*)(ws + OFF_QB);
  unsigned short* KB   = (unsigned short*)(ws + OFF_KB);
  unsigned short* VT   = (unsigned short*)(ws + OFF_VT);
  float*          ADD  = (float*)(ws + OFF_ADD);
  float*          HQ   = (float*)(ws + OFF_HQ);
  float*          HK   = (float*)(ws + OFF_HK);
  unsigned short* QP   = (unsigned short*)(ws + OFF_QP);
  unsigned short* KPT  = (unsigned short*)(ws + OFF_KPT);
  float*          KSUM = (float*)(ws + OFF_KSUM);
  float*          DEN  = (float*)(ws + OFF_DEN);
  unsigned short* KVT  = (unsigned short*)(ws + OFF_KVT);
  float*          OUT1 = (float*)(ws + OFF_OUT1);
  float*          PART = (float*)(ws + OFF_PART);

  // 1) bf16 conversions
  k_f32_to_bf16<<<4096, 256, 0, stream>>>(x, XBF, ROWS * DLEN);
  k_f32_to_bf16<<<512, 256, 0, stream>>>(Wq, WQB, DM * DLEN);
  k_f32_to_bf16<<<512, 256, 0, stream>>>(Wk, WKB, DM * DLEN);
  k_f32_to_bf16<<<512, 256, 0, stream>>>(Wv, WVB, DM * DLEN);
  k_f32_to_bf16<<<512, 256, 0, stream>>>(Wadd, WAB, DM * DLEN);
  k_f32_to_bf16<<<256, 256, 0, stream>>>(rf, RFB, MF * DM);

  // 2) fused QKV + residual projection (WMMA)
  k_qkv<<<dim3(DM / 16, ROWS / 16), 32, 0, stream>>>(XBF, WQB, WKB, WVB, WAB,
                                                     QB, KB, VT, ADD);
  // 3) row norms
  k_norm<<<dim3(ROWS / 8, 2), 256, 0, stream>>>(QB, KB, HQ, HK);
  // 4) random-feature map (WMMA + sin/cos)
  k_phi<<<dim3(MF / 16, ROWS / 16, 2), 32, 0, stream>>>(QB, KB, RFB, HQ, HK,
                                                        QP, KPT);
  // 5) k' column sums
  k_ksum<<<NB * 2 * MF, 128, 0, stream>>>(KPT, KSUM);
  // 6) kv = k'^T v per batch (WMMA)
  k_kv<<<dim3(DM / 16, 2 * MF / 16, NB), 32, 0, stream>>>(KPT, VT, KVT);
  // 7) denom
  k_denom<<<ROWS / 8, 256, 0, stream>>>(QP, KSUM, DEN);
  // 8) num/denom + residual (WMMA)
  k_out1<<<dim3(DM / 16, ROWS / 16), 32, 0, stream>>>(QP, KVT, DEN, ADD,
                                                      b_add, OUT1);
  // 9) final classifier: split-K partials + reduce (deterministic, no atomics)
  k_final_partial<<<dim3(8, NC), 256, 0, stream>>>(OUT1, Wfin, PART);
  k_final_reduce<<<1, 512, 0, stream>>>(PART, b_fin, (float*)d_out);
}
